// GATNet_4432406250034
// MI455X (gfx1250) — matrix operand order repaired and checked by reading, not compiled, not run on hardware
//
#include <hip/hip_runtime.h>
#include <hip/hip_bf16.h>

// ---------------- static config ----------------
#define NN 50000
#define EE 800000
#define GG 512
#define HC 128          // HEADS*HID
#define JK 288

typedef __attribute__((ext_vector_type(16))) __bf16 v16bf;
typedef __attribute__((ext_vector_type(8)))  float  v8f;

__device__ __forceinline__ __bf16 f2bf(float f) {
    unsigned u = __float_as_uint(f);
    unsigned r = (u + 0x7FFFu + ((u >> 16) & 1u)) >> 16;
    unsigned short h = (unsigned short)r;
    __bf16 out; __builtin_memcpy(&out, &h, 2); return out;
}

// monotone float<->uint mapping for atomicMax on floats (handles negatives)
__device__ __forceinline__ unsigned mapf(float f) {
    unsigned u = __float_as_uint(f);
    return (u & 0x80000000u) ? ~u : (u | 0x80000000u);
}
__device__ __forceinline__ float unmapf(unsigned m) {
    unsigned v = (m & 0x80000000u) ? (m & 0x7FFFFFFFu) : ~m;
    return __uint_as_float(v);
}

// ---------------- utility fills ----------------
__global__ void fill_f(float* p, float v, long long n) {
    long long t = blockIdx.x * (long long)blockDim.x + threadIdx.x;
    if (t < n) p[t] = v;
}
__global__ void fill_u(unsigned* p, unsigned v, long long n) {
    long long t = blockIdx.x * (long long)blockDim.x + threadIdx.x;
    if (t < n) p[t] = v;
}

// ---------------- bf16 conversions ----------------
// out[n*K+k] = bf16(xsrc[n*ld+k] + vrow[k])   (virtual-node broadcast add fused)
__global__ void convert_x_bf16(const float* __restrict__ xsrc, int ld,
                               const float* __restrict__ vrow,
                               __bf16* __restrict__ out, int n_rows, int K) {
    long long t = blockIdx.x * (long long)blockDim.x + threadIdx.x;
    if (t >= (long long)n_rows * K) return;
    int n = (int)(t / K), k = (int)(t % K);
    out[t] = f2bf(xsrc[(size_t)n * ld + k] + vrow[k]);
}

// Wt[o*K+k] = bf16(W[k*O+o])   (weights transposed so B-fragment loads are contiguous)
__global__ void transpose_w_bf16(const float* __restrict__ W, __bf16* __restrict__ Wt,
                                 int K, int O) {
    int t = blockIdx.x * blockDim.x + threadIdx.x;
    if (t >= K * O) return;
    int o = t / K, k = t % K;
    Wt[t] = f2bf(W[(size_t)k * O + o]);
}

// ---------------- WMMA GEMM: C[M,128] = A[M,K](bf16) x B[K,128](bf16, stored as Bt[128,K]) ----------------
// block = 256 threads = 8 waves; each wave computes a 16x16 tile; block covers 16 rows x 128 cols.
__global__ void wmma_gemm_bf16(const __bf16* __restrict__ A, const __bf16* __restrict__ Bt,
                               float* __restrict__ C, int K, int Nc) {
    int wave = threadIdx.x >> 5;
    int lane = threadIdx.x & 31;
    int nBase = wave * 16;
    int mBase = blockIdx.x * 16;
    if (nBase >= Nc) return;                 // wave-uniform; EXEC stays all-1s for WMMA
    int r = lane & 15, half = lane >> 4;

    const __bf16* arow = A  + (size_t)(mBase + r) * K;
    const __bf16* brow = Bt + (size_t)(nBase + r) * K;

    v8f acc = {};
    for (int k0 = 0; k0 < K; k0 += 32) {
        v16bf a, b;
        // A 16x32 fragment: lanes0-15 rows M, elems 0..7 -> K=k0+half*8.., elems 8..15 -> K=k0+16+half*8..
        #pragma unroll
        for (int j = 0; j < 8; ++j) {
            a[j]     = arow[k0 + half * 8 + j];
            a[8 + j] = arow[k0 + 16 + half * 8 + j];
        }
        // B 32x16 fragment: lane=N; elems 0..7 -> K=k0+half*8.., elems 8..15 -> K=k0+16+half*8..
        #pragma unroll
        for (int j = 0; j < 8; ++j) {
            b[j]     = brow[k0 + half * 8 + j];
            b[8 + j] = brow[k0 + 16 + half * 8 + j];
        }
        acc = __builtin_amdgcn_wmma_f32_16x16x32_bf16(false, a, false, b,
                                                      (short)0, acc, false, false);
    }
    // C/D layout: VGPR v -> M = v + 8*half ; lane%16 -> N
    int c = lane & 15;
    #pragma unroll
    for (int v = 0; v < 8; ++v)
        C[(size_t)(mBase + v + 8 * half) * Nc + nBase + c] = acc[v];
}

// ---------------- attention pieces ----------------
// w_e[k*4+h] = sum_c W_edge[k,h*32+c] * att_edge[h,c]   (folds edge GEMM into a 16x4 matrix)
__global__ void edge_w(const float* __restrict__ We, const float* __restrict__ atte,
                       float* __restrict__ w_e) {
    int t = threadIdx.x;
    if (t < 64) {
        int k = t >> 2, h = t & 3;
        float s = 0.f;
        for (int c = 0; c < 32; ++c) s += We[k * HC + h * 32 + c] * atte[h * 32 + c];
        w_e[t] = s;
    }
}

// alpha_src/dst per (node, head)
__global__ void node_alpha(const float* __restrict__ xh, const float* __restrict__ atts,
                           const float* __restrict__ attd, float* __restrict__ as_,
                           float* __restrict__ ad_, int n_rows) {
    int t = blockIdx.x * blockDim.x + threadIdx.x;
    if (t >= n_rows * 4) return;
    int n = t >> 2, h = t & 3;
    const float* row = xh + (size_t)n * HC + h * 32;
    float ss = 0.f, sd = 0.f;
    for (int c = 0; c < 32; ++c) { float v = row[c]; ss += v * atts[h * 32 + c]; sd += v * attd[h * 32 + c]; }
    as_[t] = ss; ad_[t] = sd;
}

// pass 1: logits + leaky-relu + segment max (via mapped uint atomicMax)
__global__ void edge_logits(const int* __restrict__ src, const int* __restrict__ dst,
                            const float* __restrict__ ea, const float* __restrict__ w_e,
                            const float* __restrict__ as_, const float* __restrict__ ad_,
                            float* __restrict__ abuf, unsigned* __restrict__ amaxU) {
    int e = blockIdx.x * blockDim.x + threadIdx.x;
    if (e >= EE) return;
    int s = src[e], d = dst[e];
    float f[16];
    const float* er = ea + (size_t)e * 16;
    #pragma unroll
    for (int k = 0; k < 16; ++k) f[k] = er[k];
    #pragma unroll
    for (int h = 0; h < 4; ++h) {
        float a = as_[s * 4 + h] + ad_[d * 4 + h];
        float aeh = 0.f;
        #pragma unroll
        for (int k = 0; k < 16; ++k) aeh += f[k] * w_e[k * 4 + h];
        a += aeh;
        a = a > 0.f ? a : 0.2f * a;          // leaky_relu(0.2)
        abuf[(size_t)e * 4 + h] = a;
        atomicMax(&amaxU[d * 4 + h], mapf(a));
    }
}

// pass 2: exp(a - max) + denominator
__global__ void edge_exp(const int* __restrict__ dst, float* __restrict__ abuf,
                         const unsigned* __restrict__ amaxU, float* __restrict__ den) {
    int t = blockIdx.x * blockDim.x + threadIdx.x;
    if (t >= EE * 4) return;
    int e = t >> 2, h = t & 3;
    int d = dst[e];
    float ex = expf(abuf[t] - unmapf(amaxU[d * 4 + h]));
    abuf[t] = ex;
    atomicAdd(&den[d * 4 + h], ex);
}

// pass 3: weighted message scatter: msg[dst,c] += (ex/den) * xh[src,c]
__global__ void edge_msg(const int* __restrict__ src, const int* __restrict__ dst,
                         const float* __restrict__ abuf, const float* __restrict__ den,
                         const float* __restrict__ xh, float* __restrict__ msg) {
    long long t = blockIdx.x * (long long)blockDim.x + threadIdx.x;
    if (t >= (long long)EE * HC) return;
    int e = (int)(t >> 7);
    int c = (int)(t & (HC - 1));
    int h = c >> 5;
    int s = src[e], d = dst[e];
    float dn = den[d * 4 + h]; dn = dn > 1e-16f ? dn : 1e-16f;
    float w = abuf[(size_t)e * 4 + h] / dn;
    atomicAdd(&msg[(size_t)d * HC + c], w * xh[(size_t)s * HC + c]);
}

// bias + (optional head-mean) + layer-norm + relu -> jk[:, off:off+feat]; warp per node
__global__ void node_post(const float* __restrict__ msg, const float* __restrict__ bias,
                          const float* __restrict__ g, const float* __restrict__ b,
                          float* __restrict__ jk, int off, int feat, int meanHeads) {
    int warp = threadIdx.x >> 5, lane = threadIdx.x & 31;
    int n = blockIdx.x * 8 + warp;
    if (n >= NN) return;
    const float* mrow = msg + (size_t)n * HC;
    int per = feat >> 5;                     // 4 (feat=128) or 1 (feat=32)
    float v[4]; float s = 0.f, q = 0.f;
    for (int i = 0; i < per; ++i) {
        int c = lane + i * 32;
        float x;
        if (meanHeads) x = 0.25f * (mrow[c] + mrow[c + 32] + mrow[c + 64] + mrow[c + 96]);
        else           x = mrow[c];
        x += bias[c];
        v[i] = x; s += x; q += x * x;
    }
    for (int m = 16; m > 0; m >>= 1) { s += __shfl_xor(s, m, 32); q += __shfl_xor(q, m, 32); }
    float mean = s / feat;
    float var  = q / feat - mean * mean;
    float inv  = rsqrtf(var + 1e-5f);
    for (int i = 0; i < per; ++i) {
        int c = lane + i * 32;
        float y = (v[i] - mean) * inv * g[c] + b[c];
        jk[(size_t)n * JK + off + c] = y > 0.f ? y : 0.f;
    }
}

// global add pool over batch
__global__ void pool_scatter(const float* __restrict__ jk, const int* __restrict__ batch,
                             float* __restrict__ pool) {
    long long t = blockIdx.x * (long long)blockDim.x + threadIdx.x;
    if (t >= (long long)NN * JK) return;
    int n = (int)(t / JK), c = (int)(t % JK);
    atomicAdd(&pool[(size_t)batch[n] * JK + c], jk[t]);
}

// small dense: C[r,o] = sum_k A[r,k]*W[k,o] + b[o]
__global__ void gemm_rows(const float* __restrict__ A, const float* __restrict__ W,
                          const float* __restrict__ bb, float* __restrict__ C,
                          int rows, int K, int O) {
    int t = blockIdx.x * blockDim.x + threadIdx.x;
    if (t >= rows * O) return;
    int r = t / O, o = t % O;
    const float* ar = A + (size_t)r * K;
    float s = bb ? bb[o] : 0.f;
    for (int k = 0; k < K; ++k) s += ar[k] * W[(size_t)k * O + o];
    C[t] = s;
}

// in-place row layer-norm + relu; block per row, feat <= 256
__global__ void row_ln_relu(float* __restrict__ X, const float* __restrict__ g,
                            const float* __restrict__ b, int feat) {
    __shared__ float ss[256], sq[256];
    int r = blockIdx.x, t = threadIdx.x;
    float x = (t < feat) ? X[(size_t)r * feat + t] : 0.f;
    ss[t] = x; sq[t] = x * x;
    __syncthreads();
    for (int s = 128; s > 0; s >>= 1) {
        if (t < s) { ss[t] += ss[t + s]; sq[t] += sq[t + s]; }
        __syncthreads();
    }
    float mean = ss[0] / feat;
    float var  = sq[0] / feat - mean * mean;
    if (t < feat) {
        float y = (x - mean) * rsqrtf(var + 1e-5f) * g[t] + b[t];
        X[(size_t)r * feat + t] = y > 0.f ? y : 0.f;
    }
}

// ---------------- host orchestration ----------------
extern "C" void kernel_launch(void* const* d_in, const int* in_sizes, int n_in,
                              void* d_out, int out_size, void* d_ws, size_t ws_size,
                              hipStream_t stream) {
    (void)in_sizes; (void)n_in; (void)out_size; (void)ws_size;
    auto F = [&](int i) { return (const float*)d_in[i]; };

    const float* x     = F(0);
    const int*   src   = (const int*)d_in[1];
    const int*   dst   = src + EE;
    const float* ea    = F(2);
    const int*   batch = (const int*)d_in[3];
    // params flattened: top-level insertion order; nested dicts alphabetical (jax pytree)
    const float* embW = F(5);  const float* embB = F(6);
    const int gatBase[3] = {7, 15, 23};   // W, W_edge, att_dst, att_edge, att_src, bias, ln_b, ln_g
    const float* lin0W = F(31); const float* lin0b = F(32); const float* lin0lb = F(33); const float* lin0lg = F(34);
    const float* lin1W = F(35); const float* lin1b = F(36); const float* lin1lb = F(37); const float* lin1lg = F(38);
    const float* outW  = F(39); const float* outb  = F(40);
    const float* vemb[3] = {F(41), F(42), F(43)};

    // workspace bump allocator
    char* wp = (char*)d_ws;
    auto alloc = [&](size_t bytes) { void* r = wp; wp += (bytes + 255) & ~(size_t)255; return r; };
    __bf16*   xbf   = (__bf16*)  alloc((size_t)NN * HC * 2);
    __bf16*   wtbf  = (__bf16*)  alloc((size_t)HC * HC * 2);
    float*    w_e   = (float*)   alloc(64 * 4);
    float*    xh    = (float*)   alloc((size_t)NN * HC * 4);
    float*    asrc  = (float*)   alloc((size_t)NN * 4 * 4);
    float*    adst  = (float*)   alloc((size_t)NN * 4 * 4);
    float*    abuf  = (float*)   alloc((size_t)EE * 4 * 4);
    unsigned* amaxU = (unsigned*)alloc((size_t)NN * 4 * 4);
    float*    den   = (float*)   alloc((size_t)NN * 4 * 4);
    float*    msg   = (float*)   alloc((size_t)NN * HC * 4);
    float*    jk    = (float*)   alloc((size_t)NN * JK * 4);
    float*    pool  = (float*)   alloc((size_t)GG * JK * 4);
    float*    t0    = (float*)   alloc((size_t)GG * 256 * 4);
    float*    t1    = (float*)   alloc((size_t)GG * 128 * 4);
    float*    t2    = (float*)   alloc((size_t)GG * 128 * 4);

    const int  Kl[3]   = {64, 128, 128};
    const int  featL[3] = {128, 128, 32};
    const int  offL[3]  = {0, 128, 256};
    const unsigned NEG_INF_MAPPED = 0x007FFFFFu;   // mapf(-inf)

    auto blocks = [](long long n, int bs) { return (unsigned)((n + bs - 1) / bs); };

    for (int i = 0; i < 3; ++i) {
        int K = Kl[i], base = gatBase[i];
        const float* W    = F(base + 0);
        const float* We   = F(base + 1);
        const float* attd = F(base + 2);
        const float* atte = F(base + 3);
        const float* atts = F(base + 4);
        const float* bias = F(base + 5);
        const float* lnb  = F(base + 6);
        const float* lng  = F(base + 7);

        const float* xsrc = (i == 0) ? x : (jk + offL[i - 1]);
        int ld = (i == 0) ? 64 : JK;

        transpose_w_bf16<<<blocks(K * HC, 256), 256, 0, stream>>>(W, wtbf, K, HC);
        convert_x_bf16<<<blocks((long long)NN * K, 256), 256, 0, stream>>>(xsrc, ld, vemb[i], xbf, NN, K);
        wmma_gemm_bf16<<<NN / 16, 256, 0, stream>>>(xbf, wtbf, xh, K, HC);

        edge_w<<<1, 64, 0, stream>>>(We, atte, w_e);
        node_alpha<<<blocks(NN * 4, 256), 256, 0, stream>>>(xh, atts, attd, asrc, adst, NN);

        fill_u<<<blocks(NN * 4, 256), 256, 0, stream>>>(amaxU, NEG_INF_MAPPED, NN * 4);
        fill_f<<<blocks(NN * 4, 256), 256, 0, stream>>>(den, 0.f, NN * 4);
        fill_f<<<blocks((long long)NN * HC, 256), 256, 0, stream>>>(msg, 0.f, (long long)NN * HC);

        edge_logits<<<blocks(EE, 256), 256, 0, stream>>>(src, dst, ea, w_e, asrc, adst, abuf, amaxU);
        edge_exp<<<blocks(EE * 4, 256), 256, 0, stream>>>(dst, abuf, amaxU, den);
        edge_msg<<<blocks((long long)EE * HC, 256), 256, 0, stream>>>(src, dst, abuf, den, xh, msg);

        node_post<<<NN / 8, 256, 0, stream>>>(msg, bias, lng, lnb, jk, offL[i], featL[i], i == 2 ? 1 : 0);
    }

    // global add pool + head MLP
    fill_f<<<blocks(GG * JK, 256), 256, 0, stream>>>(pool, 0.f, GG * JK);
    pool_scatter<<<blocks((long long)NN * JK, 256), 256, 0, stream>>>(jk, batch, pool);

    gemm_rows<<<blocks(GG * 256, 256), 256, 0, stream>>>(pool, lin0W, lin0b, t0, GG, JK, 256);
    row_ln_relu<<<GG, 256, 0, stream>>>(t0, lin0lg, lin0lb, 256);
    gemm_rows<<<blocks(GG * 128, 256), 256, 0, stream>>>(t0, lin1W, lin1b, t1, GG, 256, 128);
    row_ln_relu<<<GG, 256, 0, stream>>>(t1, lin1lg, lin1lb, 128);
    gemm_rows<<<blocks(GG * 128, 256), 256, 0, stream>>>(t1, embW, embB, t2, GG, 128, 128);
    gemm_rows<<<blocks(GG * 12, 256), 256, 0, stream>>>(t2, outW, outb, (float*)d_out, GG, 128, 12);
}
